// GNN_GCNConv_homogen_basic_2723009265694
// MI455X (gfx1250) — matrix operand order repaired and checked by reading, not compiled, not run on hardware
//
#include <hip/hip_runtime.h>

typedef __attribute__((ext_vector_type(16))) __bf16 v16bf;
typedef __attribute__((ext_vector_type(8)))  float  v8f;

#define F   128
#define FV4 32   // F/4

// ---------------------------------------------------------------------------
// K1: prep — fuse W1@Wc into bf16 transposed WfT[o][i], transpose B_bil to
// bf16 BT[g][f], and compute fused bias b1@Wc.
// ---------------------------------------------------------------------------
__global__ void prep_weights(const float* __restrict__ W1, const float* __restrict__ b1,
                             const float* __restrict__ Wc, const float* __restrict__ Bbil,
                             __bf16* __restrict__ WfT, __bf16* __restrict__ BT,
                             float* __restrict__ bfused) {
    int idx = blockIdx.x * blockDim.x + threadIdx.x;
    if (idx < F * F) {
        int o = idx >> 7, i = idx & 127;
        float s = 0.f;
        for (int j = 0; j < F; ++j) s += W1[i * F + j] * Wc[j * F + o];
        WfT[o * F + i] = (__bf16)s;
    } else if (idx < 2 * F * F) {
        int t = idx - F * F;
        int g = t >> 7, f = t & 127;
        BT[g * F + f] = (__bf16)Bbil[f * F + g];
    } else if (idx < 2 * F * F + F) {
        int o = idx - 2 * F * F;
        float s = 0.f;
        for (int j = 0; j < F; ++j) s += b1[j] * Wc[j * F + o];
        bfused[o] = s;
    }
}

// ---------------------------------------------------------------------------
// WMMA GEMM: C[M,128] = A[M,128] (fp32, cvt->bf16) @ BT^T + bias
// BT is [128(out) x 128(in)] bf16 (pre-transposed), staged in LDS.
// One wave -> 16 rows x 128 cols: 8 accumulators, 4 K-chunks of 32,
// 32 v_wmma_f32_16x16x32_bf16 per wave with A reused across 8 N-tiles.
// ---------------------------------------------------------------------------
__global__ void __launch_bounds__(256)
gemm_nt128(const float* __restrict__ A, const __bf16* __restrict__ BT,
           const float* __restrict__ bias, float* __restrict__ C, int M) {
    __shared__ __align__(32) __bf16 sB[F * F];   // 32 KB
    {
        const uint4* src = (const uint4*)BT;
        uint4* dst = (uint4*)sB;
        for (int i = threadIdx.x; i < (F * F * 2) / 16; i += 256) dst[i] = src[i];
    }
    __syncthreads();

    const int wave = threadIdx.x >> 5;
    const int lane = threadIdx.x & 31;
    const int half = lane >> 4;       // lane group 0: lanes 0-15, 1: lanes 16-31
    const int lrow = lane & 15;
    const int m0 = (blockIdx.x * 8 + wave) * 16;
    if (m0 >= M) return;              // wave-uniform: EXEC stays all-1s

    v8f acc[8] = {};
    const float* arow = A + (size_t)(m0 + lrow) * F;

    for (int kc = 0; kc < 4; ++kc) {
        const int k0 = kc * 32;
        // k order used here (B matches): half 0 holds K {k0..k0+15},
        //                  half 1 holds K {k0+16..k0+31}
        const float4* p0 = (const float4*)(arow + k0 + half * 16);
        float4 f0 = p0[0], f1 = p0[1];
        const float4* p1 = (const float4*)(arow + k0 + half * 16 + 8);
        float4 f2 = p1[0], f3 = p1[1];
        v16bf a;
        a[0]  = (__bf16)f0.x; a[1]  = (__bf16)f0.y; a[2]  = (__bf16)f0.z; a[3]  = (__bf16)f0.w;
        a[4]  = (__bf16)f1.x; a[5]  = (__bf16)f1.y; a[6]  = (__bf16)f1.z; a[7]  = (__bf16)f1.w;
        a[8]  = (__bf16)f2.x; a[9]  = (__bf16)f2.y; a[10] = (__bf16)f2.z; a[11] = (__bf16)f2.w;
        a[12] = (__bf16)f3.x; a[13] = (__bf16)f3.y; a[14] = (__bf16)f3.z; a[15] = (__bf16)f3.w;

        #pragma unroll
        for (int nt = 0; nt < 8; ++nt) {
            // B layout: lane holds column nt*16+lrow, K = k0 + half*16 + 0..15
            // (contiguous 32B in LDS, 32B aligned -> two ds_load_b128)
            const v16bf* bp = (const v16bf*)(sB + (nt * 16 + lrow) * F + k0 + half * 16);
            v16bf b = *bp;
            acc[nt] = __builtin_amdgcn_wmma_f32_16x16x32_bf16(
                false, a, false, b, (short)0, acc[nt], false, false);
        }
    }

    // D layout: VGPR r, lane<16 -> (M=r, N=lane); lane>=16 -> (M=r+8, N=lane-16)
    #pragma unroll
    for (int nt = 0; nt < 8; ++nt) {
        const int col = nt * 16 + lrow;
        const float bv = bias ? bias[col] : 0.0f;
        #pragma unroll
        for (int r = 0; r < 8; ++r) {
            const int row = m0 + half * 8 + r;
            C[(size_t)row * F + col] = acc[nt][r] + bv;
        }
    }
}

// ---------------------------------------------------------------------------
// Degree / norm kernels
// ---------------------------------------------------------------------------
__global__ void deg_init(float* __restrict__ deg, int N) {
    int i = blockIdx.x * blockDim.x + threadIdx.x;
    if (i < N) deg[i] = 1.0f;                       // self-loop
}

__global__ void deg_accum(const long long* __restrict__ col, float* __restrict__ deg, int E) {
    int e = blockIdx.x * blockDim.x + threadIdx.x;
    if (e < E) atomicAdd(&deg[(int)col[e]], 1.0f);
}

__global__ void deg_to_dis(float* __restrict__ deg, int N) {
    int i = blockIdx.x * blockDim.x + threadIdx.x;
    if (i < N) {
        float d = deg[i];
        deg[i] = (d > 0.f) ? rsqrtf(d) : 0.f;       // in-place: deg -> dis
    }
}

// agg[i] = h[i]*dis[i]^2 + bc   (self-loop term + GCN bias)
__global__ void agg_init(const float* __restrict__ h, const float* __restrict__ dis,
                         const float* __restrict__ bc, float* __restrict__ agg, int N) {
    int t = blockIdx.x * blockDim.x + threadIdx.x;
    if (t >= N * FV4) return;
    int n = t >> 5, l = t & 31;
    float d = dis[n];
    float w = d * d;
    float4 hv = ((const float4*)h)[(size_t)n * FV4 + l];
    float4 bv = ((const float4*)bc)[l];
    float4 o;
    o.x = hv.x * w + bv.x; o.y = hv.y * w + bv.y;
    o.z = hv.z * w + bv.z; o.w = hv.w * w + bv.w;
    ((float4*)agg)[(size_t)n * FV4 + l] = o;
}

// agg[col] += h[row] * dis[row]*dis[col]   — one (edge, 4-feature) per thread
__global__ void scatter_edges(const long long* __restrict__ ei, const float* __restrict__ h,
                              const float* __restrict__ dis, float* __restrict__ agg, int E) {
    long long t = (long long)blockIdx.x * blockDim.x + threadIdx.x;
    if (t >= (long long)E * FV4) return;
    int e = (int)(t >> 5), l = (int)(t & 31);
    int r = (int)ei[e];          // source j
    int c = (int)ei[E + e];      // target i
    float nrm = dis[r] * dis[c];
    float4 hv = ((const float4*)h)[(size_t)r * FV4 + l];
    float* dst = agg + (size_t)c * F + l * 4;
    atomicAdd(dst + 0, hv.x * nrm);
    atomicAdd(dst + 1, hv.y * nrm);
    atomicAdd(dst + 2, hv.z * nrm);
    atomicAdd(dst + 3, hv.w * nrm);
}

// score[e] = dot(outB[src_e], out[dst_e]) + b_bil   — one wave32 per edge
__global__ void edge_score(const long long* __restrict__ ei, const float* __restrict__ outB,
                           const float* __restrict__ out, const float* __restrict__ bbil,
                           float* __restrict__ score, int E) {
    int w = blockIdx.x * (blockDim.x >> 5) + (threadIdx.x >> 5);
    int l = threadIdx.x & 31;
    if (w >= E) return;
    int s = (int)ei[w];
    int d = (int)ei[E + w];
    float4 a = ((const float4*)outB)[(size_t)s * FV4 + l];
    float4 b = ((const float4*)out )[(size_t)d * FV4 + l];
    float p = a.x * b.x + a.y * b.y + a.z * b.z + a.w * b.w;
    #pragma unroll
    for (int off = 16; off > 0; off >>= 1) p += __shfl_down(p, off, 32);
    if (l == 0) score[w] = p + bbil[0];
}

// ---------------------------------------------------------------------------
extern "C" void kernel_launch(void* const* d_in, const int* in_sizes, int n_in,
                              void* d_out, int out_size, void* d_ws, size_t ws_size,
                              hipStream_t stream) {
    const float*     x     = (const float*)d_in[0];
    const long long* edge  = (const long long*)d_in[1];   // int64 [2,E]
    const long long* pedge = (const long long*)d_in[2];   // int64 [2,Ep]
    const float*     W1    = (const float*)d_in[3];
    const float*     b1    = (const float*)d_in[4];
    const float*     Wc    = (const float*)d_in[5];
    const float*     bc    = (const float*)d_in[6];
    const float*     Bb    = (const float*)d_in[7];       // [1,128,128]
    const float*     bb    = (const float*)d_in[8];       // [1]

    const int N  = in_sizes[0] / F;    // 100000
    const int E  = in_sizes[1] / 2;    // 500000
    const int Ep = in_sizes[2] / 2;    // 500000

    char* ws = (char*)d_ws;
    const size_t hBytes = (size_t)N * F * sizeof(float);          // 51.2 MB
    float*  h    = (float*)ws;                                    // h, later outB
    float*  agg  = (float*)(ws + hBytes);                         // agg / out
    float*  deg  = (float*)(ws + 2 * hBytes);                     // deg -> dis
    const size_t degBytes = (((size_t)N * sizeof(float)) + 255) & ~(size_t)255;
    __bf16* WfT  = (__bf16*)(ws + 2 * hBytes + degBytes);         // 32 KB
    __bf16* BT   = WfT + F * F;                                   // 32 KB
    float*  bfus = (float*)(BT + F * F);                          // 512 B

    const int gemmBlocks = ((N + 15) / 16 + 7) / 8;

    prep_weights<<<(2 * F * F + F + 255) / 256, 256, 0, stream>>>(W1, b1, Wc, Bb, WfT, BT, bfus);
    gemm_nt128 <<<gemmBlocks, 256, 0, stream>>>(x, WfT, bfus, h, N);
    deg_init   <<<(N + 255) / 256, 256, 0, stream>>>(deg, N);
    deg_accum  <<<(Ep + 255) / 256, 256, 0, stream>>>(pedge + Ep, deg, Ep);
    deg_to_dis <<<(N + 255) / 256, 256, 0, stream>>>(deg, N);
    agg_init   <<<((size_t)N * FV4 + 255) / 256, 256, 0, stream>>>(h, deg, bc, agg, N);
    scatter_edges<<<(int)(((size_t)Ep * FV4 + 255) / 256), 256, 0, stream>>>(pedge, h, deg, agg, Ep);
    gemm_nt128 <<<gemmBlocks, 256, 0, stream>>>(agg, BT, nullptr, h, N);     // h := outB
    edge_score <<<(E + 7) / 8, 256, 0, stream>>>(edge, h, agg, bb, (float*)d_out, E);
}
